// ChiENNLayer_86139864089507
// MI455X (gfx1250) — matrix operand order repaired and checked by reading, not compiled, not run on hardware
//
#include <hip/hip_runtime.h>
#include <hip/hip_bf16.h>

typedef __attribute__((ext_vector_type(16))) _Float16 v16h;
typedef __attribute__((ext_vector_type(8)))  _Float16 h8;
typedef __attribute__((ext_vector_type(8)))  float    v8f;
typedef __attribute__((ext_vector_type(4)))  float    f4;

#define D   128
#define CS  6
#define KK  3
#define SWS 136   // padded LDS row stride in halves: 272B = 68 dwords, gcd(68,64)=4

union Frag { v16h v; h8 h[2]; };

__device__ __forceinline__ float elu_f(float x) {
    return x > 0.f ? x : (__expf(x) - 1.f);
}

// num_neighbors (may be negative; callers clamp where needed)
__device__ __forceinline__ int node_nn(const int* __restrict__ cidx, int node) {
    const int* ci = cidx + node * CS;
    int pad = 0;
    #pragma unroll
    for (int j = 0; j < CS; ++j) pad += (ci[j] == -1) ? 1 : 0;
    return (pad == CS) ? 0 : (4 - pad);
}

// ---------------------------------------------------------------------------
// Kernel 1: convert+transpose weights to f16 (WT[m][n][k] = W_m[k][n]),
// m: 0..2 = Wk[0..2], 3 = Wf, 4 = Ws, 5 = Wp.  Also zero sentinel row 0 of E*.
// ---------------------------------------------------------------------------
__global__ void prep_kernel(const float* __restrict__ Wk, const float* __restrict__ Wf,
                            const float* __restrict__ Ws, const float* __restrict__ Wp,
                            _Float16* __restrict__ WT,
                            _Float16* __restrict__ E0, _Float16* __restrict__ E1,
                            _Float16* __restrict__ E2) {
    int t = blockIdx.x * blockDim.x + threadIdx.x;
    if (t < 3 * D) {
        _Float16* E = (t < D) ? E0 : ((t < 2 * D) ? E1 : E2);
        E[t % D] = (_Float16)0.f;
    }
    if (t >= 6 * D * D) return;
    int m   = t / (D * D);
    int rem = t % (D * D);
    int n   = rem / D;
    int k   = rem % D;
    const float* W;
    switch (m) {
        case 0: W = Wk;             break;
        case 1: W = Wk + D * D;     break;
        case 2: W = Wk + 2 * D * D; break;
        case 3: W = Wf;             break;
        case 4: W = Ws;             break;
        default: W = Wp;            break;
    }
    WT[t] = (_Float16)W[k * D + n];
}

// Cooperative fill of one 128x128 f16 matrix into padded LDS (256 threads).
__device__ __forceinline__ void fill_sW(_Float16* sW, const _Float16* __restrict__ Wb,
                                        int tid) {
    #pragma unroll
    for (int it = 0; it < 8; ++it) {
        const int ch  = it * 256 + tid;    // 2048 chunks of 8 halves
        const int row = ch >> 4;
        const int kc  = ch & 15;
        *(h8*)(sW + row * SWS + kc * 8) = *(const h8*)(Wb + row * D + kc * 8);
    }
}

// ---------------------------------------------------------------------------
// Kernel 2: one wave per 16-row strip of x; 8 waves/block share LDS-staged
// weights (one matrix at a time, om loop fully unrolled).  Outputs are padded
// to whole strips, so every store is unconditional (OOB waves duplicate the
// last strip -> benign identical writes).  All five outputs stored f16
// (accumulation stays f32 in the WMMA accumulators):
//   E_i = f16(x @ Wk[i] + bk[i])   (rows shifted by +1; row 0 is zeros)
//   S   = f16(x @ Ws + bs),  P = f16(x @ Wp + bp)
// ---------------------------------------------------------------------------
__global__ void __launch_bounds__(256)
stage1_kernel(const float* __restrict__ x,
              const _Float16* __restrict__ WT,
              const float* __restrict__ bk,
              const float* __restrict__ bs,
              const float* __restrict__ bp,
              _Float16* __restrict__ E0, _Float16* __restrict__ E1,
              _Float16* __restrict__ E2,
              _Float16* __restrict__ S, _Float16* __restrict__ P, int N) {
    __shared__ __align__(16) _Float16 sW[D * SWS];

    const int tid    = threadIdx.x;
    const int lane   = tid & 31;
    const int strips = (N + 15) >> 4;
    int wave = blockIdx.x * 8 + (tid >> 5);
    if (wave >= strips) wave = strips - 1;         // duplicate last strip (benign)
    const int r0   = wave << 4;
    const int half = lane >> 4;
    const int lr   = lane & 15;

    int arow = r0 + lr;
    if (arow >= N) arow = N - 1;                   // clamp reads; padded stores absorb

    // A fragments (16x32 f16 per K-step), f32 -> f16 in registers.
    Frag a[4];
    #pragma unroll
    for (int kt = 0; kt < 4; ++kt) {
        const float* xp = x + (size_t)arow * D + kt * 32 + half * 8;
        f4 f0 = *(const f4*)(xp);
        f4 f1 = *(const f4*)(xp + 4);
        f4 f2 = *(const f4*)(xp + 16);
        f4 f3 = *(const f4*)(xp + 20);
        h8 lo, hi;
        #pragma unroll
        for (int j = 0; j < 4; ++j) {
            lo[j]     = (_Float16)f0[j];
            lo[4 + j] = (_Float16)f1[j];
            hi[j]     = (_Float16)f2[j];
            hi[4 + j] = (_Float16)f3[j];
        }
        a[kt].h[0] = lo;
        a[kt].h[1] = hi;
    }

    _Float16* outs[5] = { E0, E1, E2, S, P };
    const int rbase = r0 + half * 8;               // first of this lane's 8 rows

    #pragma unroll
    for (int om = 0; om < 5; ++om) {
        const int wm = (om < 3) ? om : om + 1;     // skip Wf (m=3) here
        fill_sW(sW, WT + (size_t)wm * D * D, tid);
        __syncthreads();

        const float* bias = (om < 3) ? (bk + om * D) : ((om == 3) ? bs : bp);
        const int rshift = (om < 3) ? 1 : 0;       // E has sentinel zero row 0
        #pragma unroll
        for (int nt = 0; nt < 8; ++nt) {
            const int col = nt * 16 + lr;
            const float bv = bias[col];
            v8f c = { bv, bv, bv, bv, bv, bv, bv, bv };
            #pragma unroll
            for (int kt = 0; kt < 4; ++kt) {
                Frag b;
                const _Float16* wp = sW + col * SWS + kt * 32 + half * 8;
                b.h[0] = *(const h8*)(wp);
                b.h[1] = *(const h8*)(wp + 16);
                c = __builtin_amdgcn_wmma_f32_16x16x32_f16(
                        false, a[kt].v, false, b.v, (short)0, c, false, false);
            }
            _Float16* Op = outs[om] + (size_t)(rbase + rshift) * D + col;
            #pragma unroll
            for (int r = 0; r < 8; ++r) Op[r * D] = (_Float16)c[r];
        }
        __syncthreads();
    }
}

// ---------------------------------------------------------------------------
// Kernel 3: one wave per 4 nodes (16 msg rows); 8 waves/block share LDS WfT.
// Fully branchless: node indices clamped for reads; OOB output rows are
// redirected to a dump row in the workspace via pointer select.
// ---------------------------------------------------------------------------
__global__ void __launch_bounds__(256)
stage2_kernel(const int* __restrict__ cidx,
              const int* __restrict__ pidx,
              const _Float16* __restrict__ WT,
              const float* __restrict__ bf,
              const _Float16* __restrict__ E0,
              const _Float16* __restrict__ E1,
              const _Float16* __restrict__ E2,
              const _Float16* __restrict__ S, const _Float16* __restrict__ P,
              float* __restrict__ dump,
              float* __restrict__ out, int N) {
    __shared__ __align__(16) _Float16 sW[D * SWS];

    const int tid = threadIdx.x;
    fill_sW(sW, WT + (size_t)3 * D * D, tid);      // WfT
    __syncthreads();

    const int lane    = tid & 31;
    const int nstrips = (N + 3) >> 2;
    int wave = blockIdx.x * 8 + (tid >> 5);
    if (wave >= nstrips) wave = nstrips - 1;       // duplicate last strip (benign)
    const int nb    = wave << 2;                   // first of 4 nodes
    const int half  = lane >> 4;
    const int lr    = lane & 15;
    const int node  = nb + (lr >> 2);              // node of this lane's A row
    const int nodeR = node < N ? node : N - 1;     // clamped for reads
    const int c     = lr & 3;                      // msg slot of this lane's A row

    const int nn  = node_nn(cidx, nodeR);
    const int* ci = cidx + (size_t)nodeR * CS;
    int g[KK];
    #pragma unroll
    for (int i = 0; i < KK; ++i) g[i] = (nn > i) ? (ci[c + i] + 1) : 0;  // row 0 = zeros
    const bool rowvalid = (c + 1 <= nn);

    const _Float16* Es[3] = { E0, E1, E2 };

    Frag a[4];
    #pragma unroll
    for (int kt = 0; kt < 4; ++kt) {
        const int o = kt * 32 + half * 8;
        float acc0[8], acc1[8];
        #pragma unroll
        for (int j = 0; j < 8; ++j) { acc0[j] = 0.f; acc1[j] = 0.f; }
        #pragma unroll
        for (int i = 0; i < KK; ++i) {
            const _Float16* er = Es[i] + (size_t)g[i] * D;
            h8 e0 = *(const h8*)(er + o);
            h8 e1 = *(const h8*)(er + o + 16);
            #pragma unroll
            for (int j = 0; j < 8; ++j) { acc0[j] += (float)e0[j]; acc1[j] += (float)e1[j]; }
        }
        h8 lo, hi;
        #pragma unroll
        for (int j = 0; j < 8; ++j) {
            lo[j] = (_Float16)(rowvalid ? elu_f(acc0[j]) : 0.f);
            hi[j] = (_Float16)(rowvalid ? elu_f(acc1[j]) : 0.f);
        }
        a[kt].h[0] = lo;
        a[kt].h[1] = hi;
    }

    // Epilogue node assignment from C-fragment layout:
    // VGPR r: lanes 0-15 hold msg-row r (nodes nb, nb+1), lanes 16-31 row r+8.
    const int node_lo = nb + 2 * half;
    const int node_hi = node_lo + 1;
    const int nlR = node_lo < N ? node_lo : N - 1;
    const int nhR = node_hi < N ? node_hi : N - 1;

    int t_lo = node_nn(cidx, nlR); const int cnt_lo = t_lo < 0 ? 0 : t_lo;
    int t_hi = node_nn(cidx, nhR); const int cnt_hi = t_hi < 0 ? 0 : t_hi;
    const int pl = pidx[nlR];
    const int ph = pidx[nhR];

    const _Float16* Srow_lo = S + (size_t)nlR * D;
    const _Float16* Srow_hi = S + (size_t)nhR * D;
    const _Float16* Prow_lo = P + (size_t)pl * D;
    const _Float16* Prow_hi = P + (size_t)ph * D;
    float* Orow_lo = (node_lo < N) ? (out + (size_t)node_lo * D) : dump;
    float* Orow_hi = (node_hi < N) ? (out + (size_t)node_hi * D) : dump;

    #pragma unroll
    for (int nt = 0; nt < 8; ++nt) {
        const int col = nt * 16 + lr;
        v8f cacc = { 0.f, 0.f, 0.f, 0.f, 0.f, 0.f, 0.f, 0.f };
        #pragma unroll
        for (int kt = 0; kt < 4; ++kt) {
            Frag b;
            const _Float16* wp = sW + col * SWS + kt * 32 + half * 8;
            b.h[0] = *(const h8*)(wp);
            b.h[1] = *(const h8*)(wp + 16);
            cacc = __builtin_amdgcn_wmma_f32_16x16x32_f16(
                       false, a[kt].v, false, b.v, (short)0, cacc, false, false);
        }
        const float slo = cacc[0] + cacc[1] + cacc[2] + cacc[3]; // rows c=0..3, node_lo
        const float shi = cacc[4] + cacc[5] + cacc[6] + cacc[7]; // rows c=0..3, node_hi
        const float bfv = bf[col];
        const float vlo = slo + (float)cnt_lo * bfv
                        + (float)Srow_lo[col] + (float)Prow_lo[col];
        const float vhi = shi + (float)cnt_hi * bfv
                        + (float)Srow_hi[col] + (float)Prow_hi[col];
        Orow_lo[col] = elu_f(vlo);
        Orow_hi[col] = elu_f(vhi);
    }
}

// ---------------------------------------------------------------------------
extern "C" void kernel_launch(void* const* d_in, const int* in_sizes, int n_in,
                              void* d_out, int out_size, void* d_ws, size_t ws_size,
                              hipStream_t stream) {
    (void)n_in; (void)out_size; (void)ws_size;
    const float* x    = (const float*)d_in[0];
    const int*   cidx = (const int*)d_in[1];
    const int*   pidx = (const int*)d_in[2];
    const float* Wk   = (const float*)d_in[3];
    const float* bk   = (const float*)d_in[4];
    const float* Wf   = (const float*)d_in[5];
    const float* bf   = (const float*)d_in[6];
    const float* Ws   = (const float*)d_in[7];
    const float* bs   = (const float*)d_in[8];
    const float* Wp   = (const float*)d_in[9];
    const float* bp   = (const float*)d_in[10];
    const int N  = in_sizes[0] / D;
    const int Nr = ((N + 15) / 16) * 16;           // rows padded to whole strips

    char*  ws  = (char*)d_ws;
    size_t off = 0;
    _Float16* WT = (_Float16*)(ws + off);
    off += (size_t)6 * D * D * sizeof(_Float16);
    off = (off + 255) & ~(size_t)255;
    const size_t Esz = (size_t)(Nr + 1) * D * sizeof(_Float16);
    _Float16* E0 = (_Float16*)(ws + off); off += Esz; off = (off + 255) & ~(size_t)255;
    _Float16* E1 = (_Float16*)(ws + off); off += Esz; off = (off + 255) & ~(size_t)255;
    _Float16* E2 = (_Float16*)(ws + off); off += Esz; off = (off + 255) & ~(size_t)255;
    _Float16* S = (_Float16*)(ws + off); off += (size_t)Nr * D * sizeof(_Float16);
    off = (off + 255) & ~(size_t)255;
    _Float16* P = (_Float16*)(ws + off); off += (size_t)Nr * D * sizeof(_Float16);
    off = (off + 255) & ~(size_t)255;
    float* dump = (float*)(ws + off); off += (size_t)D * sizeof(float);

    {
        const int total = 6 * D * D;
        prep_kernel<<<(total + 255) / 256, 256, 0, stream>>>(Wk, Wf, Ws, Wp, WT, E0, E1, E2);
    }
    {
        const int strips = (N + 15) / 16;  // one wave per strip, 8 waves/block
        const int blocks = (strips + 7) / 8;
        stage1_kernel<<<blocks, 256, 0, stream>>>(x, WT, bk, bs, bp, E0, E1, E2, S, P, N);
    }
    {
        const int strips = (N + 3) / 4;    // one wave per 4 nodes, 8 waves/block
        const int blocks = (strips + 7) / 8;
        stage2_kernel<<<blocks, 256, 0, stream>>>(cidx, pidx, WT, bf, E0, E1, E2, S, P,
                                                  dump, (float*)d_out, N);
    }
}
